// RNN_test_8650064134447
// MI455X (gfx1250) — matrix operand order repaired and checked by reading, not compiled, not run on hardware
//
#include <hip/hip_runtime.h>

#define B_   256
#define S_   4096
#define HID  128
#define G4   512          // 4*HID gate columns
#define CH   64           // input prefetch chunk length (steps)

typedef __attribute__((ext_vector_type(16))) _Float16 v16h;
typedef __attribute__((ext_vector_type(8)))  _Float16 v8h;
typedef __attribute__((ext_vector_type(8)))  float    v8f;

__device__ __forceinline__ float sigf(float x) {
    // 1/(1+e^-x) via exp2/rcp hardware trans ops
    return __builtin_amdgcn_rcpf(1.0f + __builtin_amdgcn_exp2f(-1.44269504f * x));
}
__device__ __forceinline__ float tanh_(float x) {
    // tanh(x) = 1 - 2/(e^{2x}+1);  e^{2x} = exp2(2.885390082*x)
    float e = __builtin_amdgcn_exp2f(2.885390082f * x);
    return 1.0f - 2.0f * __builtin_amdgcn_rcpf(e + 1.0f);
}

// ---- prologue 1: W_eff(512,3) = W_ih @ W_inp ; b_eff = W_ih@b_inp + b_ih + b_hh
__global__ void k_build_eff(const float* __restrict__ W_inp, const float* __restrict__ b_inp,
                            const float* __restrict__ W_ih,  const float* __restrict__ b_ih,
                            const float* __restrict__ b_hh,  float* __restrict__ weff) {
    int j = blockIdx.x * blockDim.x + threadIdx.x;
    if (j >= G4) return;
    float w0 = 0.f, w1 = 0.f, w2 = 0.f, bb = 0.f;
    for (int k = 0; k < 64; ++k) {
        float a = W_ih[j * 64 + k];
        w0 += a * W_inp[k * 3 + 0];
        w1 += a * W_inp[k * 3 + 1];
        w2 += a * W_inp[k * 3 + 2];
        bb += a * b_inp[k];
    }
    bb += b_ih[j] + b_hh[j];
    weff[j * 4 + 0] = w0; weff[j * 4 + 1] = w1;
    weff[j * 4 + 2] = w2; weff[j * 4 + 3] = bb;
}

// ---- prologue 2: W_hh f32 -> f16 (row-major 512x128)
__global__ void k_cvt_whh(const float* __restrict__ W_hh, _Float16* __restrict__ whh16) {
    int i = blockIdx.x * blockDim.x + threadIdx.x;
    if (i < G4 * HID) whh16[i] = (_Float16)W_hh[i];
}

// ---- main persistent LSTM kernel: 16 blocks x 256 threads (8 wave32 / WGP)
//  block = 16-batch tile; wave w owns gate cols {w*16..} of each of i/f/g/o.
__global__ __launch_bounds__(256) void k_lstm(
    const float* __restrict__ inputs, const float* __restrict__ hx0,
    const float* __restrict__ cx0,    const float* __restrict__ W_out,
    const float* __restrict__ b_out,  const float* __restrict__ weff,
    const _Float16* __restrict__ whh16, float* __restrict__ out)
{
    __shared__ _Float16 hxbuf[2][16 * 136];      // ping-pong hx (f16), padded stride 136
    __shared__ float    inpC[2][CH][16][4];      // prefetched input chunks (double buffer)
    __shared__ float    hxf32[16][132];          // epilogue only

    const int tid  = threadIdx.x;
    const int wave = tid >> 5;
    const int lane = tid & 31;
    const int hi   = lane >> 4;     // half-wave (C/D rows r vs r+8)
    const int n    = lane & 15;     // N column within 16-wide tile
    const int wb   = blockIdx.x * 16;
    const int h    = wave * 16 + n; // hidden index this lane produces

    // input-staging mapping: thread stages 12 floats (=4 steps) of one batch
    const int sb = tid >> 4;        // batch staged by this thread
    const int sq = tid & 15;        // 48-byte block within the batch's chunk
    const float* sbase = inputs + (size_t)(wb + sb) * (S_ * 3) + sq * 12;

    // B fragments of W_hh^T (loop-invariant, register resident).
    // B 32x16 f16, k order used here (same as A): lanes 0-15 hold K 0..15 of col n, lanes 16-31 hold K 16..31.
    v16h bfrag[4][4];
    #pragma unroll
    for (int g = 0; g < 4; ++g) {
        const int row = g * 128 + h;                 // W_hh row == gate column
        #pragma unroll
        for (int kf = 0; kf < 4; ++kf)
            bfrag[g][kf] = *(const v16h*)(whh16 + row * HID + kf * 32 + hi * 16);
    }

    // Per-lane folded input weights: (w0,w1,w2,b_eff) for this lane's 4 gate cols.
    float4 wg[4];
    #pragma unroll
    for (int g = 0; g < 4; ++g) {
        const float* p = weff + (g * 128 + h) * 4;
        wg[g] = make_float4(p[0], p[1], p[2], p[3]);
    }

    // Init state: cx in regs (accumulator layout), hx0 -> f16 buffer 0.
    float cxv[8], hxv[8];
    #pragma unroll
    for (int r = 0; r < 8; ++r) {
        const int b = r + 8 * hi;
        cxv[r] = cx0[(wb + b) * HID + h];
        float h0 = hx0[(wb + b) * HID + h];
        hxv[r] = h0;
        hxbuf[0][b * 136 + h] = (_Float16)h0;
    }

    // Stage chunk 0 (steps 0..63) into buffer 0.
    {
        const float4* p = (const float4*)sbase;    // 16B-aligned (sq*48 + batch*48KB)
        float4 a0 = p[0], a1 = p[1], a2 = p[2];
        float* d = &inpC[0][sq * 4][sb][0];        // row stride = 16*4 floats
        d[0]   = a0.x; d[1]   = a0.y; d[2]   = a0.z;
        d[64]  = a0.w; d[65]  = a1.x; d[66]  = a1.y;
        d[128] = a1.z; d[129] = a1.w; d[130] = a2.x;
        d[192] = a2.y; d[193] = a2.z; d[194] = a2.w;
    }
    __syncthreads();

    #pragma unroll 1
    for (int t = 0; t < S_; ++t) {
        const int cur = t & 1, nxt = cur ^ 1;
        const int cb  = (t >> 6) & 1;              // input chunk buffer in use

        // Issue next-chunk global loads FIRST so latency hides under this step.
        const bool doStage = ((t & (CH - 1)) == 0) && (t + CH < S_);
        float4 a0, a1, a2;
        if (doStage) {
            const float4* p = (const float4*)(sbase + (t + CH) * 3);  // 16B-aligned
            a0 = p[0]; a1 = p[1]; a2 = p[2];
        }

        // A fragments (hx 16x128 -> 4 frags of 16x32 f16).
        // A k order used here (same as B): lane<16 K 0..15; lane>=16 K 16..31.
        v16h afrag[4];
        const _Float16* abase = &hxbuf[cur][n * 136];
        #pragma unroll
        for (int kf = 0; kf < 4; ++kf) {
            v8h lo = *(const v8h*)(abase + kf * 32 + hi * 16);
            v8h hh = *(const v8h*)(abase + kf * 32 + hi * 16 + 8);
            afrag[kf] = __builtin_shufflevector(lo, hh,
                0,1,2,3,4,5,6,7,8,9,10,11,12,13,14,15);
        }

        // C init = exact f32 input projection: inp3 . W_eff + b_eff
        v8f acc[4];
        #pragma unroll
        for (int r = 0; r < 8; ++r) {
            const float4 ip = *(const float4*)&inpC[cb][t & (CH - 1)][8 * hi + r][0];
            #pragma unroll
            for (int g = 0; g < 4; ++g)
                acc[g][r] = fmaf(ip.x, wg[g].x,
                            fmaf(ip.y, wg[g].y,
                            fmaf(ip.z, wg[g].z, wg[g].w)));
        }

        // gates += hx @ W_hh^T   (16 x v_wmma_f32_16x16x32_f16 per wave)
        #pragma unroll
        for (int g = 0; g < 4; ++g) {
            #pragma unroll
            for (int kf = 0; kf < 4; ++kf)
                acc[g] = __builtin_amdgcn_wmma_f32_16x16x32_f16(
                    false, afrag[kf], false, bfrag[g][kf],
                    (short)0, acc[g], false, false);
        }

        // Commit staged chunk into the other buffer (readers start only after
        // the chunk-boundary barrier, so this is race-free).
        if (doStage) {
            float* d = &inpC[cb ^ 1][sq * 4][sb][0];
            d[0]   = a0.x; d[1]   = a0.y; d[2]   = a0.z;
            d[64]  = a0.w; d[65]  = a1.x; d[66]  = a1.y;
            d[128] = a1.z; d[129] = a1.w; d[130] = a2.x;
            d[192] = a2.y; d[193] = a2.z; d[194] = a2.w;
        }

        // Pointwise LSTM cell. i/f/g/o for (b,h) sit in the same lane slot
        // across acc[0..3] — no cross-wave gate exchange needed.
        #pragma unroll
        for (int r = 0; r < 8; ++r) {
            const float iv = sigf(acc[0][r]);
            const float fv = sigf(acc[1][r]);
            const float gv = tanh_(acc[2][r]);
            const float ov = sigf(acc[3][r]);
            const float c2 = fv * cxv[r] + iv * gv;
            const float h2 = ov * tanh_(c2);
            cxv[r] = c2;
            hxv[r] = h2;
            // reference's jnp.all(hx > hx_new) carry branch is never taken:
            // carry = hx_new.
            hxbuf[nxt][(r + 8 * hi) * 136 + h] = (_Float16)h2;
        }
        __syncthreads();  // dscnt-waited barrier: covers hx ping-pong + chunk handoff
    }

    // ---- epilogue: out = hx_last @ W_out^T + b_out (f32, exact hx regs)
    #pragma unroll
    for (int r = 0; r < 8; ++r)
        hxf32[r + 8 * hi][h] = hxv[r];
    __syncthreads();

    for (int idx = tid; idx < 16 * 40; idx += 256) {
        const int b = idx / 40, jo = idx - b * 40;
        const float* wrow = W_out + jo * HID;
        float s = b_out[jo];
        #pragma unroll 4
        for (int k = 0; k < HID; k += 4)
            s += hxf32[b][k+0]*wrow[k+0] + hxf32[b][k+1]*wrow[k+1]
               + hxf32[b][k+2]*wrow[k+2] + hxf32[b][k+3]*wrow[k+3];
        out[(wb + b) * 40 + jo] = s;
    }
}

extern "C" void kernel_launch(void* const* d_in, const int* in_sizes, int n_in,
                              void* d_out, int out_size, void* d_ws, size_t ws_size,
                              hipStream_t stream)
{
    const float* inputs = (const float*)d_in[0];
    const float* hx0    = (const float*)d_in[1];
    const float* cx0    = (const float*)d_in[2];
    const float* W_inp  = (const float*)d_in[3];
    const float* b_inp  = (const float*)d_in[4];
    const float* W_ih   = (const float*)d_in[5];
    const float* b_ih   = (const float*)d_in[6];
    const float* W_hh   = (const float*)d_in[7];
    const float* b_hh   = (const float*)d_in[8];
    const float* W_out  = (const float*)d_in[9];
    const float* b_out  = (const float*)d_in[10];
    float* out = (float*)d_out;

    // workspace: [0,128KB) W_hh f16 ; [128KB, 136KB) W_eff/b_eff packed float4
    _Float16* whh16 = (_Float16*)d_ws;
    float*    weff  = (float*)((char*)d_ws + (size_t)G4 * HID * sizeof(_Float16));

    k_cvt_whh <<<(G4 * HID) / 256, 256, 0, stream>>>(W_hh, whh16);
    k_build_eff<<<2, 256, 0, stream>>>(W_inp, b_inp, W_ih, b_ih, b_hh, weff);
    k_lstm<<<16, 256, 0, stream>>>(inputs, hx0, cx0, W_out, b_out, weff, whh16, out);
}
